// _BinPackingSeq2SeqDecoder_17609365914443
// MI455X (gfx1250) — matrix operand order repaired and checked by reading, not compiled, not run on hardware
//
#include <hip/hip_runtime.h>
#include <hip/hip_bf16.h>

// ---------------------------------------------------------------------------
// LSTM decoder: B=256, H=1024, OUT=6, T=128
// Per step: gates = [x|h] @ Wcat^T (+bias) -> activations -> c,h update
//           box = h @ W_fc^T + b_fc ; x_next = rint(box @ W_th^T + b_th)
//
// gfx1250 paths used:
//   * v_wmma_f32_16x16x32_bf16 (fp32 accumulation)
//   * global_load_async_to_lds_b128 + s_wait_asynccnt (double-buffered B tile)
//   * ds_load_b128 broadcast of the staged B tile to all 16 waves
// Weight matrix staged once per step from L2 (16.8 MB bf16, L2-resident).
// ---------------------------------------------------------------------------

typedef __bf16 bf16_t;
typedef __attribute__((ext_vector_type(16))) __bf16 v16bf;
typedef __attribute__((ext_vector_type(8)))  __bf16 v8bf;
typedef __attribute__((ext_vector_type(8)))  float  v8f;

#define HID   1024
#define BATCH 256
#define NOUT  6
#define SEQ   128
#define KDIM  2048   // x(1024) ++ h(1024)
#define NGATE 4096   // 4*HID
#define NCHUNK (KDIM / 32)   // 64 k-chunks of 32

#if __has_builtin(__builtin_amdgcn_s_wait_asynccnt)
#define WAIT_ASYNC(n) __builtin_amdgcn_s_wait_asynccnt(n)
#else
#define WAIT_ASYNC(n) asm volatile("s_wait_asynccnt %0" :: "i"(n))
#endif

__device__ __forceinline__ void async_b128_to_lds(unsigned int lds_addr,
                                                  unsigned long long gaddr) {
    // GLOBAL_LOAD_ASYNC_TO_LDS_B128 (GV mode): 16B per active lane, ASYNCcnt.
    asm volatile("global_load_async_to_lds_b128 %0, %1, off"
                 :: "v"(lds_addr), "v"(gaddr) : "memory");
}

__device__ __forceinline__ float sigmoidf_(float x) {
    return 1.0f / (1.0f + __expf(-x));
}

__device__ __forceinline__ bf16_t to_bf16(float f) { return (bf16_t)f; }

// ---------------------------------------------------------------------------
// One-time: Wcat[g][0:1024] = W_ih[g][:], Wcat[g][1024:2048] = W_hh[g][:]
// bsum[g] = b_ih[g] + b_hh[g]
// ---------------------------------------------------------------------------
__global__ __launch_bounds__(256) void prep_weights_kernel(
    const float* __restrict__ Wih, const float* __restrict__ Whh,
    const float* __restrict__ bih, const float* __restrict__ bhh,
    bf16_t* __restrict__ Wcat, float* __restrict__ bsum)
{
    int g = blockIdx.x;                       // 0..4095 gate row
    for (int k = threadIdx.x; k < HID; k += 256) {
        Wcat[g * KDIM + k]        = to_bf16(Wih[g * HID + k]);
        Wcat[g * KDIM + HID + k]  = to_bf16(Whh[g * HID + k]);
    }
    if (threadIdx.x == 0) bsum[g] = bih[g] + bhh[g];
}

// ---------------------------------------------------------------------------
// One-time: c0 = z, h0 = z (bf16 into xh), x0 = 0
// ---------------------------------------------------------------------------
__global__ __launch_bounds__(256) void init_state_kernel(
    const float* __restrict__ z, float* __restrict__ cbuf,
    bf16_t* __restrict__ xh0)
{
    int b = blockIdx.x;
#pragma unroll
    for (int i = 0; i < 4; ++i) {
        int j = threadIdx.x + i * 256;
        float v = z[b * HID + j];
        cbuf[b * HID + j]          = v;
        xh0[b * KDIM + j]          = to_bf16(0.0f);   // x part
        xh0[b * KDIM + HID + j]    = to_bf16(v);      // h part
    }
}

// ---------------------------------------------------------------------------
// Per-step heavy kernel: gates GEMM (M=256,N=4096,K=2048) + LSTM cell update.
//
// grid = 64 blocks (one j-tile of 16 hidden units), 512 threads = 16 waves.
// Wave w owns batch rows [16w, 16w+16) and computes all four gates for the
// block's j-tile (4 wmma accumulators -> i,f,g,o coexist per lane).
//
// B tile (4 gate rows x 16 j x 32 k bf16 = 4KB/chunk) is async-staged into a
// double-buffered LDS tile by waves 0..7 (one b128 per lane per chunk) and
// read by every wave via ds_load_b128 -> Wcat streams from L2 exactly once
// per step. A fragments (xh rows) are register double-buffered one chunk
// ahead so global loads overlap the wmma+ds phase.
// ---------------------------------------------------------------------------
__global__ __launch_bounds__(512) void lstm_gates_kernel(
    const bf16_t* __restrict__ xh_in,   // [256][2048] (x_t ++ h_t)
    bf16_t* __restrict__ xh_out,        // [256][2048] h_{t+1} -> cols 1024..2047
    const bf16_t* __restrict__ Wcat,    // [4096][2048]
    const float* __restrict__ bsum,     // [4096]
    float* __restrict__ cbuf,           // [256][1024]
    float* __restrict__ hbuf)           // [256][1024] h_{t+1} fp32 (for fc)
{
    __shared__ __align__(32) unsigned char ldsB[2][4096];  // [buf][row*64 + kbyte]

    const int tid  = threadIdx.x;
    const int wave = tid >> 5;          // 0..15 = m_tile
    const int lane = tid & 31;
    const int l    = lane & 15;
    const int hi   = lane >> 4;

    const int j0 = blockIdx.x * 16;     // j-tile of this block
    const int m0 = wave * 16;           // batch tile of this wave

    // --- B staging setup (threads 0..255: one b128 lane each per chunk) ---
    const int q    = tid;               // staging lane id
    const int srow = q >> 2;            // 0..63 : gate*16 + jl
    const int sseg = q & 3;             // 16B segment within 64B k-row
    const int sgt  = srow >> 4;
    const int sjl  = srow & 15;
    const unsigned long long gstage =
        (unsigned long long)(const void*)(Wcat + (sgt * HID + j0 + sjl) * KDIM)
        + (unsigned long long)(sseg * 16);
    const unsigned int lds_base = (unsigned int)(unsigned long long)&ldsB[0][0];
    const unsigned int lstage   = lds_base + (unsigned int)(srow * 64 + sseg * 16);

    // --- accumulators: acc[gt].f[r] = gates[m0 + r + 8*hi][j0 + l] ---
    union AccU { v8f v; float f[8]; };
    AccU acc[4];
#pragma unroll
    for (int gt = 0; gt < 4; ++gt)
#pragma unroll
        for (int r = 0; r < 8; ++r) acc[gt].f[r] = 0.0f;

    // A fragment source (contiguous k order, B uses the same):
    // lane<16 holds K {kc..kc+15}, lane>=16 {kc+16..kc+31}
    const bf16_t* arow = xh_in + (m0 + l) * KDIM + hi * 16;

    // prologue: stage chunk 0, preload A chunk 0
    if (tid < 256) async_b128_to_lds(lstage, gstage);

    union AU { v16bf v; v8bf h8[2]; };
    AU a_cur;
    {
        const v8bf* pa = (const v8bf*)(arow);
        a_cur.h8[0] = pa[0];
        a_cur.h8[1] = pa[1];
    }

    for (int ic = 0; ic < NCHUNK; ++ic) {
        const int buf = ic & 1;

        // issue async stage of next chunk into the other buffer
        if (ic + 1 < NCHUNK) {
            if (tid < 256) {
                async_b128_to_lds(lstage + (unsigned int)((buf ^ 1) * 4096),
                                  gstage + (unsigned long long)((ic + 1) * 64));
            }
        }
        // wait for *current* chunk's async data (in-order completion per wave)
        if (tid < 256) {
            if (ic + 1 < NCHUNK) WAIT_ASYNC(1); else WAIT_ASYNC(0);
        }
        __syncthreads();

        // prefetch next A fragment while wmma consumes the current one
        AU a_nxt;
        if (ic + 1 < NCHUNK) {
            const v8bf* pa = (const v8bf*)(arow + (ic + 1) * 32);
            a_nxt.h8[0] = pa[0];
            a_nxt.h8[1] = pa[1];
        }

        // B fragments from LDS: lane (l,hi) reads row gt*16+l, bytes hi*32..+31
#pragma unroll
        for (int gt = 0; gt < 4; ++gt) {
            const v16bf b = *(const v16bf*)&ldsB[buf][(gt * 16 + l) * 64 + hi * 32];
            acc[gt].v = __builtin_amdgcn_wmma_f32_16x16x32_bf16(
                false, a_cur.v, false, b, (short)0, acc[gt].v, false, false);
        }
        a_cur = a_nxt;

        __syncthreads();   // all reads of buf done before it is restaged
    }

    // ---- LSTM cell update (register-resident i,f,g,o per lane) ----
    const int j   = j0 + l;
    const float bi  = bsum[0 * HID + j];
    const float bff = bsum[1 * HID + j];
    const float bg  = bsum[2 * HID + j];
    const float bo  = bsum[3 * HID + j];
#pragma unroll
    for (int r = 0; r < 8; ++r) {
        const int m = m0 + r + 8 * hi;
        const float ig = sigmoidf_(acc[0].f[r] + bi);
        const float fg = sigmoidf_(acc[1].f[r] + bff);
        const float gg = tanhf(acc[2].f[r] + bg);
        const float og = sigmoidf_(acc[3].f[r] + bo);
        const float cold = cbuf[m * HID + j];
        const float cn = fg * cold + ig * gg;
        const float hn = og * tanhf(cn);
        cbuf[m * HID + j] = cn;
        hbuf[m * HID + j] = hn;
        xh_out[m * KDIM + HID + j] = to_bf16(hn);
    }
}

// ---------------------------------------------------------------------------
// Per-step tiny kernel: box = h @ W_fc^T + b_fc  (written to d_out[b,t,:])
//                       x_next = rint(box @ W_th^T + b_th)  -> xh_out x-part
// One block per batch row.
// ---------------------------------------------------------------------------
__global__ __launch_bounds__(256) void lstm_out_kernel(
    const float* __restrict__ hbuf,   // [256][1024]
    const float* __restrict__ Wfc,    // [6][1024]
    const float* __restrict__ bfc,    // [6]
    const float* __restrict__ Wth,    // [1024][6]
    const float* __restrict__ bth,    // [1024]
    float* __restrict__ out,          // [256][128][6]
    bf16_t* __restrict__ xh_out,      // x part: cols 0..1023
    int t)
{
    __shared__ float red[NOUT][256];
    __shared__ float box[NOUT];
    const int b   = blockIdx.x;
    const int tid = threadIdx.x;

    float part[NOUT];
#pragma unroll
    for (int o = 0; o < NOUT; ++o) part[o] = 0.0f;

    const float* hrow = hbuf + b * HID;
    for (int k = tid; k < HID; k += 256) {
        const float hv = hrow[k];
#pragma unroll
        for (int o = 0; o < NOUT; ++o) part[o] += hv * Wfc[o * HID + k];
    }
#pragma unroll
    for (int o = 0; o < NOUT; ++o) red[o][tid] = part[o];
    __syncthreads();
    for (int s = 128; s > 0; s >>= 1) {
        if (tid < s) {
#pragma unroll
            for (int o = 0; o < NOUT; ++o) red[o][tid] += red[o][tid + s];
        }
        __syncthreads();
    }
    if (tid < NOUT) {
        const float v = red[tid][0] + bfc[tid];
        box[tid] = v;
        out[(b * SEQ + t) * NOUT + tid] = v;
    }
    __syncthreads();

#pragma unroll
    for (int i = 0; i < 4; ++i) {
        const int j = tid + i * 256;
        float v = bth[j];
#pragma unroll
        for (int o = 0; o < NOUT; ++o) v += box[o] * Wth[j * NOUT + o];
        xh_out[b * KDIM + j] = to_bf16(rintf(v));   // jnp.round = half-to-even
    }
}

// ---------------------------------------------------------------------------
// Launch: prep weights once, init state, then 128 x (gates, out).
// Workspace layout (bytes):
//   [0, 16M)        Wcat bf16 [4096][2048]
//   +0x1000000      xh ping  bf16 [256][2048]
//   +0x1100000      xh pong  bf16 [256][2048]
//   +0x1200000      c fp32   [256][1024]
//   +0x1300000      h fp32   [256][1024]
//   +0x1400000      bsum fp32 [4096]
// ---------------------------------------------------------------------------
extern "C" void kernel_launch(void* const* d_in, const int* in_sizes, int n_in,
                              void* d_out, int out_size, void* d_ws, size_t ws_size,
                              hipStream_t stream) {
    const float* z    = (const float*)d_in[0];
    const float* W_ih = (const float*)d_in[1];
    const float* W_hh = (const float*)d_in[2];
    const float* b_ih = (const float*)d_in[3];
    const float* b_hh = (const float*)d_in[4];
    const float* W_fc = (const float*)d_in[5];
    const float* b_fc = (const float*)d_in[6];
    const float* W_th = (const float*)d_in[7];
    const float* b_th = (const float*)d_in[8];
    // d_in[9] = seq_len (device scalar); fixed at 128 per reference setup.

    char* ws = (char*)d_ws;
    bf16_t* Wcat = (bf16_t*)(ws);
    bf16_t* xh0  = (bf16_t*)(ws + 0x1000000);
    bf16_t* xh1  = (bf16_t*)(ws + 0x1100000);
    float*  cbuf = (float*)(ws + 0x1200000);
    float*  hbuf = (float*)(ws + 0x1300000);
    float*  bsum = (float*)(ws + 0x1400000);
    float*  out  = (float*)d_out;

    prep_weights_kernel<<<NGATE, 256, 0, stream>>>(W_ih, W_hh, b_ih, b_hh, Wcat, bsum);
    init_state_kernel<<<BATCH, 256, 0, stream>>>(z, cbuf, xh0);

    for (int t = 0; t < SEQ; ++t) {
        bf16_t* xin  = (t & 1) ? xh1 : xh0;
        bf16_t* xout = (t & 1) ? xh0 : xh1;
        lstm_gates_kernel<<<64, 512, 0, stream>>>(xin, xout, Wcat, bsum, cbuf, hbuf);
        lstm_out_kernel<<<BATCH, 256, 0, stream>>>(hbuf, W_fc, b_fc, W_th, b_th, out, xout, t);
    }
}
